// GAT_33844342293157
// MI455X (gfx1250) — matrix operand order repaired and checked by reading, not compiled, not run on hardware
//
#include <hip/hip_runtime.h>
#include <hip/hip_bf16.h>
#include <stdint.h>

#define NEG_SLOPE 0.2f

typedef __attribute__((ext_vector_type(16))) __bf16 v16bf;
typedef __attribute__((ext_vector_type(8)))  float  v8f;

union FragU { v16bf v; uint4 q[2]; };

__device__ __forceinline__ unsigned short f2bf(float f) {
  unsigned u = __float_as_uint(f);
  unsigned r = u + 0x7fffu + ((u >> 16) & 1u);   // round-to-nearest-even
  return (unsigned short)(r >> 16);
}
// order-preserving float<->uint transform for atomicMax on signed floats
__device__ __forceinline__ unsigned encf(float f) {
  unsigned u = __float_as_uint(f);
  return (u & 0x80000000u) ? ~u : (u | 0x80000000u);
}
__device__ __forceinline__ float decf(unsigned u) {
  u = (u & 0x80000000u) ? (u & 0x7fffffffu) : ~u;
  return __uint_as_float(u);
}

// ---------------------------------------------------------------- utilities
__global__ void fill_f32(float* p, float v, int n) {
  int t = blockIdx.x * blockDim.x + threadIdx.x;
  if (t < n) p[t] = v;
}
__global__ void fill_u32(unsigned* p, unsigned v, int n) {
  int t = blockIdx.x * blockDim.x + threadIdx.x;
  if (t < n) p[t] = v;
}
__global__ void cvt_bf16(const float* __restrict__ in, unsigned short* __restrict__ out, int n) {
  int t = blockIdx.x * blockDim.x + threadIdx.x;
  if (t < n) out[t] = f2bf(in[t]);
}

// Pack fp32 weight W[K,Nc] (row-major) into WMMA B-fragment order (bf16):
// Bp[((nt*(K/32)+kt)*32 + lane)*16 + i] = bf16(W[kt*32 + (lane>>4)*16 + i, nt*16 + (lane&15)])
__global__ void pack_weight(const float* __restrict__ W, unsigned short* __restrict__ Bp,
                            int K, int Nc) {
  int t = blockIdx.x * blockDim.x + threadIdx.x;
  int K32 = K >> 5;
  int total = (Nc >> 4) * K32 * 32;
  if (t >= total) return;
  int lane = t & 31;
  int rest = t >> 5;
  int kt = rest % K32;
  int nt = rest / K32;
  int n  = nt * 16 + (lane & 15);
  int kb = kt * 32 + (lane >> 4) * 16;
  unsigned short* out = Bp + (size_t)t * 16;
#pragma unroll
  for (int i = 0; i < 16; ++i)
    out[i] = f2bf(W[(size_t)(kb + i) * Nc + n]);
}

// ---------------------------------------------------------------- WMMA GEMM
// C[z] = A(bf16)[M,K] * Bp[z] + bias[z], one 16x16 tile per wave.
// Selects one of three (B, bias, C) triples via blockIdx.z (shares A via L2).
__global__ __launch_bounds__(256) void wmma_gemm3(
    const unsigned short* __restrict__ A,
    const unsigned short* __restrict__ Bp0, const unsigned short* __restrict__ Bp1,
    const unsigned short* __restrict__ Bp2,
    const float* __restrict__ bias0, const float* __restrict__ bias1,
    const float* __restrict__ bias2,
    float* __restrict__ C0, float* __restrict__ C1, float* __restrict__ C2,
    int M, int K, int Nc) {
  const unsigned short* Bp;
  const float* bias;
  float* C;
  if (blockIdx.z == 0)      { Bp = Bp0; bias = bias0; C = C0; }
  else if (blockIdx.z == 1) { Bp = Bp1; bias = bias1; C = C1; }
  else                      { Bp = Bp2; bias = bias2; C = C2; }

  int wave = threadIdx.x >> 5;
  int lane = threadIdx.x & 31;
  int tiles_n = Nc >> 4;
  int tiles   = (M >> 4) * tiles_n;
  int tile = blockIdx.x * 8 + wave;
  if (tile >= tiles) return;           // whole-wave exit: EXEC stays all-1 for WMMA
  int mt = tile / tiles_n;
  int nt = tile - mt * tiles_n;

  int m = lane & 15;
  int g = lane >> 4;                    // half-wave group
  const unsigned short* arow  = A + (size_t)(mt * 16 + m) * K + g * 16;
  const unsigned short* bbase = Bp + ((size_t)nt * (K >> 5) * 32 + lane) * 16;

  v8f acc = {};
  int K32 = K >> 5;
  for (int kt = 0; kt < K32; ++kt) {
    FragU a, b;
    const unsigned short* pa = arow + kt * 32;
    a.q[0] = *(const uint4*)(pa);        // K = kt*32 + g*16     .. +7
    a.q[1] = *(const uint4*)(pa + 8);    // K = kt*32 + g*16 + 8 .. +7
    const unsigned short* pb = bbase + (size_t)kt * 32 * 16;
    b.q[0] = *(const uint4*)(pb);
    b.q[1] = *(const uint4*)(pb + 8);
    acc = __builtin_amdgcn_wmma_f32_16x16x32_bf16(
        false, a.v, false, b.v, (short)0, acc, false, false);
  }

  int col  = nt * 16 + (lane & 15);
  int row0 = mt * 16 + g * 8;
  float bv = bias[col];
  float* crow = C + (size_t)row0 * Nc + col;
#pragma unroll
  for (int r = 0; r < 8; ++r)
    crow[(size_t)r * Nc] = acc[r] + bv;
}

// ---------------------------------------------------------------- edge phase
// One wave per edge. LOGD = log2(D); HD = 2*D (H=2 heads).
template <int LOGD>
__global__ __launch_bounds__(256) void edge_logits(
    const float* __restrict__ fs, const float* __restrict__ fd,
    const float* __restrict__ attn, const int* __restrict__ src,
    const int* __restrict__ dst, float* __restrict__ e,
    unsigned* __restrict__ mmax, int Ne) {
  int wid  = (blockIdx.x * blockDim.x + threadIdx.x) >> 5;
  int lane = threadIdx.x & 31;
  if (wid >= Ne) return;
  int s = src[wid], d = dst[wid];
  const int HD = 2 << LOGD;
  const float* ps = fs + (size_t)s * HD;
  const float* pd = fd + (size_t)d * HD;
  float a0 = 0.f, a1 = 0.f;
  for (int j = lane; j < HD; j += 32) {
    float v = ps[j] + pd[j];
    v = v > 0.f ? v : NEG_SLOPE * v;
    float t = v * attn[j];
    if (j >> LOGD) a1 += t; else a0 += t;
  }
#pragma unroll
  for (int off = 16; off; off >>= 1) {
    a0 += __shfl_xor(a0, off, 32);
    a1 += __shfl_xor(a1, off, 32);
  }
  if (lane == 0) {
    e[(size_t)wid * 2 + 0] = a0;
    e[(size_t)wid * 2 + 1] = a1;
    atomicMax(&mmax[d * 2 + 0], encf(a0));
    atomicMax(&mmax[d * 2 + 1], encf(a1));
  }
}

__global__ void edge_exp(float* __restrict__ e, const unsigned* __restrict__ mmax,
                         float* __restrict__ den, const int* __restrict__ dst, int Ne) {
  int t = blockIdx.x * blockDim.x + threadIdx.x;
  if (t >= Ne * 2) return;
  int edge = t >> 1, h = t & 1;
  int d = dst[edge];
  float ex = expf(e[t] - decf(mmax[d * 2 + h]));
  e[t] = ex;
  atomicAdd(&den[d * 2 + h], ex);
}

template <int LOGD>
__global__ __launch_bounds__(256) void edge_agg(
    const float* __restrict__ fs, const float* __restrict__ e,
    const float* __restrict__ den, const int* __restrict__ src,
    const int* __restrict__ dst, float* __restrict__ agg, int Ne) {
  int wid  = (blockIdx.x * blockDim.x + threadIdx.x) >> 5;
  int lane = threadIdx.x & 31;
  if (wid >= Ne) return;
  int s = src[wid], d = dst[wid];
  const int HD = 2 << LOGD;
  float a0 = e[(size_t)wid * 2 + 0] / den[d * 2 + 0];
  float a1 = e[(size_t)wid * 2 + 1] / den[d * 2 + 1];
  const float* ps = fs + (size_t)s * HD;
  float* pa = agg + (size_t)d * HD;
  for (int j = lane; j < HD; j += 32)
    atomicAdd(&pa[j], ps[j] * ((j >> LOGD) ? a1 : a0));
}

// ---------------------------------------------------------------- node phases
__global__ void node_h1(const float* __restrict__ agg, const float* __restrict__ fr,
                        unsigned short* __restrict__ h1b, int total) {
  int t = blockIdx.x * blockDim.x + threadIdx.x;
  if (t >= total) return;
  float v = agg[t] + fr[t];
  v = v > 0.f ? v : 0.f;
  h1b[t] = f2bf(v);
}

// sum over heads of relu(agg2+f2r) [N,2,512] -> relu -> bf16 [N,512]
__global__ void node_final(const float* __restrict__ agg, const float* __restrict__ fr,
                           unsigned short* __restrict__ hsb, int Nn) {
  int t = blockIdx.x * blockDim.x + threadIdx.x;
  if (t >= Nn * 512) return;
  int n = t >> 9, dcol = t & 511;
  size_t base = (size_t)n * 1024;
  float v0 = agg[base + dcol] + fr[base + dcol];
  v0 = v0 > 0.f ? v0 : 0.f;
  float v1 = agg[base + 512 + dcol] + fr[base + 512 + dcol];
  v1 = v1 > 0.f ? v1 : 0.f;
  float v = v0 + v1;                     // relu(v0+v1) == v0+v1 (>=0)
  hsb[t] = f2bf(v);
}

// ---------------------------------------------------------------- launcher
extern "C" void kernel_launch(void* const* d_in, const int* in_sizes, int n_in,
                              void* d_out, int out_size, void* d_ws, size_t ws_size,
                              hipStream_t stream) {
  const float* x    = (const float*)d_in[0];
  const int*   src  = (const int*)d_in[1];
  const int*   dst  = (const int*)d_in[2];
  const float* W1s  = (const float*)d_in[3];
  const float* b1s  = (const float*)d_in[4];
  const float* W1d  = (const float*)d_in[5];
  const float* b1d  = (const float*)d_in[6];
  const float* at1  = (const float*)d_in[7];
  const float* W1r  = (const float*)d_in[8];
  const float* b1r  = (const float*)d_in[9];
  const float* W2s  = (const float*)d_in[10];
  const float* b2s  = (const float*)d_in[11];
  const float* W2d  = (const float*)d_in[12];
  const float* b2d  = (const float*)d_in[13];
  const float* at2  = (const float*)d_in[14];
  const float* W2r  = (const float*)d_in[15];
  const float* b2r  = (const float*)d_in[16];
  const float* Wp   = (const float*)d_in[17];
  const float* bp   = (const float*)d_in[18];
  float* out = (float*)d_out;

  const int Nn = in_sizes[0] / 1024;   // 10000
  const int Ne = in_sizes[1];          // 160000

  // ---- workspace layout (phase-overlaid) ----
  char* w = (char*)d_ws;
  size_t off = 0;
  auto alloc = [&](size_t bytes) -> void* {
    void* p = w + off;
    off += bytes;
    off = (off + 255) & ~(size_t)255;
    return p;
  };
  // big region: 4 * N * 1024 floats
  float* big = (float*)alloc((size_t)4 * Nn * 1024 * sizeof(float));
  float* f1s  = big;
  float* f1d  = big + (size_t)1 * Nn * 512;
  float* f1r  = big + (size_t)2 * Nn * 512;
  float* agg1 = big + (size_t)3 * Nn * 512;
  float* f2s  = big;                                   // overlays layer-1 (dead by then)
  float* f2d  = big + (size_t)1 * Nn * 1024;
  float* f2r  = big + (size_t)2 * Nn * 1024;
  float* agg2 = big + (size_t)3 * Nn * 1024;

  unsigned short* xb  = (unsigned short*)alloc((size_t)Nn * 1024 * 2);
  unsigned short* hsb = xb;                            // overlay: xb dead after GEMM1
  unsigned short* h1b = (unsigned short*)alloc((size_t)Nn * 512 * 2);
  float*    ebuf = (float*)alloc((size_t)Ne * 2 * sizeof(float));
  unsigned* mbuf = (unsigned*)alloc((size_t)Nn * 2 * sizeof(unsigned));
  float*    dbuf = (float*)alloc((size_t)Nn * 2 * sizeof(float));
  const size_t WS1 = (size_t)1024 * 512;               // elements per packed matrix
  const size_t WS2 = (size_t)512 * 1024;
  unsigned short* wp1 = (unsigned short*)alloc(3 * WS1 * 2);
  unsigned short* wp2 = (unsigned short*)alloc(3 * WS2 * 2);
  unsigned short* wpp = (unsigned short*)alloc((size_t)512 * 512 * 2);

  // ---- phase 0: conversions & weight packing ----
  int nx = Nn * 1024;
  cvt_bf16<<<(nx + 255) / 256, 256, 0, stream>>>(x, xb, nx);
  pack_weight<<<128, 256, 0, stream>>>(W1s, wp1 + 0 * WS1, 1024, 512);
  pack_weight<<<128, 256, 0, stream>>>(W1d, wp1 + 1 * WS1, 1024, 512);
  pack_weight<<<128, 256, 0, stream>>>(W1r, wp1 + 2 * WS1, 1024, 512);
  pack_weight<<<128, 256, 0, stream>>>(W2s, wp2 + 0 * WS2, 512, 1024);
  pack_weight<<<128, 256, 0, stream>>>(W2d, wp2 + 1 * WS2, 512, 1024);
  pack_weight<<<128, 256, 0, stream>>>(W2r, wp2 + 2 * WS2, 512, 1024);
  pack_weight<<<64, 256, 0, stream>>>(Wp, wpp, 512, 512);

  // ---- layer 1 GEMMs: [N,1024] x [1024,512] x3 ----
  {
    int tiles = (Nn >> 4) * (512 >> 4);
    dim3 grid((tiles + 7) / 8, 1, 3);
    wmma_gemm3<<<grid, 256, 0, stream>>>(xb, wp1, wp1 + WS1, wp1 + 2 * WS1,
                                         b1s, b1d, b1r, f1s, f1d, f1r,
                                         Nn, 1024, 512);
  }
  // ---- layer 1 edge softmax + aggregation (D=256, LOGD=8) ----
  fill_u32<<<(Nn * 2 + 255) / 256, 256, 0, stream>>>(mbuf, 0u, Nn * 2);
  fill_f32<<<(Nn * 2 + 255) / 256, 256, 0, stream>>>(dbuf, 0.f, Nn * 2);
  fill_f32<<<(Nn * 512 + 255) / 256, 256, 0, stream>>>(agg1, 0.f, Nn * 512);
  edge_logits<8><<<(Ne + 7) / 8, 256, 0, stream>>>(f1s, f1d, at1, src, dst, ebuf, mbuf, Ne);
  edge_exp<<<(Ne * 2 + 255) / 256, 256, 0, stream>>>(ebuf, mbuf, dbuf, dst, Ne);
  edge_agg<8><<<(Ne + 7) / 8, 256, 0, stream>>>(f1s, ebuf, dbuf, src, dst, agg1, Ne);
  node_h1<<<(Nn * 512 + 255) / 256, 256, 0, stream>>>(agg1, f1r, h1b, Nn * 512);

  // ---- layer 2 GEMMs: [N,512] x [512,1024] x3 ----
  {
    int tiles = (Nn >> 4) * (1024 >> 4);
    dim3 grid((tiles + 7) / 8, 1, 3);
    wmma_gemm3<<<grid, 256, 0, stream>>>(h1b, wp2, wp2 + WS2, wp2 + 2 * WS2,
                                         b2s, b2d, b2r, f2s, f2d, f2r,
                                         Nn, 512, 1024);
  }
  // ---- layer 2 edge softmax + aggregation (D=512, LOGD=9) ----
  fill_u32<<<(Nn * 2 + 255) / 256, 256, 0, stream>>>(mbuf, 0u, Nn * 2);
  fill_f32<<<(Nn * 2 + 255) / 256, 256, 0, stream>>>(dbuf, 0.f, Nn * 2);
  fill_f32<<<(Nn * 1024 + 255) / 256, 256, 0, stream>>>(agg2, 0.f, Nn * 1024);
  edge_logits<9><<<(Ne + 7) / 8, 256, 0, stream>>>(f2s, f2d, at2, src, dst, ebuf, mbuf, Ne);
  edge_exp<<<(Ne * 2 + 255) / 256, 256, 0, stream>>>(ebuf, mbuf, dbuf, dst, Ne);
  edge_agg<9><<<(Ne + 7) / 8, 256, 0, stream>>>(f2s, ebuf, dbuf, src, dst, agg2, Ne);
  node_final<<<(Nn * 512 + 255) / 256, 256, 0, stream>>>(agg2, f2r, hsb, Nn);

  // ---- projection: [N,512] x [512,512] -> d_out ----
  {
    int tiles = (Nn >> 4) * (512 >> 4);
    dim3 grid((tiles + 7) / 8, 1, 1);
    wmma_gemm3<<<grid, 256, 0, stream>>>(hsb, wpp, wpp, wpp, bp, bp, bp,
                                         out, out, out, Nn, 512, 512);
  }
}
